// PoeHmmLm_19353122636099
// MI455X (gfx1250) — compile-verified, matrix operand order checked by reading, not run on hardware
//
#include <hip/hip_runtime.h>
#include <hip/hip_bf16.h>
#include <math.h>

typedef __attribute__((ext_vector_type(16))) _Float16 v16h;
typedef __attribute__((ext_vector_type(8)))  float    v8f;

#define NB 4
#define TT 96
#define VV 4096
#define CC 64
#define HH 256
#define KW 3

// ---------------------------------------------------------------------------
// Fragment loaders for v_wmma_f32_16x16x32_f16 operand layout (wave32):
//   lane L: row/col = L%16, g = L/16; half i of the v16h maps to
//   K = 16*(i/8) + 8*g + (i%8)  -> two contiguous 8-float runs per fragment.
// ---------------------------------------------------------------------------
__device__ __forceinline__ v16h load_frag_rowmajor(const float* __restrict__ rowbase,
                                                   int kk, int g) {
  const float4* p = reinterpret_cast<const float4*>(rowbase);
  int i0 = (kk + (g << 3)) >> 2;        // halves 0..7  : k = kk + 8g + 0..7
  int i1 = (kk + 16 + (g << 3)) >> 2;   // halves 8..15 : k = kk + 16 + 8g + 0..7
  float4 a0 = p[i0], a1 = p[i0 + 1], a2 = p[i1], a3 = p[i1 + 1];
  v16h f;
  f[0]  = (_Float16)a0.x; f[1]  = (_Float16)a0.y; f[2]  = (_Float16)a0.z; f[3]  = (_Float16)a0.w;
  f[4]  = (_Float16)a1.x; f[5]  = (_Float16)a1.y; f[6]  = (_Float16)a1.z; f[7]  = (_Float16)a1.w;
  f[8]  = (_Float16)a2.x; f[9]  = (_Float16)a2.y; f[10] = (_Float16)a2.z; f[11] = (_Float16)a2.w;
  f[12] = (_Float16)a3.x; f[13] = (_Float16)a3.y; f[14] = (_Float16)a3.z; f[15] = (_Float16)a3.w;
  return f;
}

__device__ __forceinline__ v16h load_fragB_kmajor(const float* __restrict__ B,
                                                  int Nn, int col, int kk, int g) {
  v16h f;
#pragma unroll
  for (int i = 0; i < 16; ++i) {
    int k = kk + ((i >> 3) << 4) + (g << 3) + (i & 7);
    f[i] = (_Float16)B[(size_t)k * Nn + col];
  }
  return f;
}

// ---------------------------------------------------------------------------
// Generic WMMA GEMM: C(MxN) = epilogue( A(MxK,row-major) @ B )
//   TRANSB==0: B row-major (KxN).  TRANSB==1: B is (NxK) row-major.
//   Epilogue: +bias[col], ReLU, +resid[row][col] (each optional, runtime,
//   hoisted out of the per-element store loop).
// One wave per 16x16 tile, K unrolled by 64 (two WMMAs / iter).
// Requires M%16==0, N%16==0, K%64==0.
// ---------------------------------------------------------------------------
template <int TRANSB>
__global__ void wmma_gemm_kernel(const float* __restrict__ A,
                                 const float* __restrict__ B,
                                 const float* __restrict__ bias,
                                 const float* __restrict__ resid,
                                 float* __restrict__ C,
                                 int M, int Nn, int Kk, int doRelu) {
  int wave = threadIdx.x >> 5;
  int lane = threadIdx.x & 31;
  int tilesN = Nn >> 4;
  int totalTiles = (M >> 4) * tilesN;
  int tile = blockIdx.x * (blockDim.x >> 5) + wave;
  if (tile >= totalTiles) return;              // wave-uniform: EXEC stays all-1s
  int ti = tile / tilesN, tj = tile % tilesN;
  int g  = lane >> 4;
  int mr = lane & 15;
  int row0 = ti << 4, col0 = tj << 4;
  int col  = col0 + mr;

  const float* Arow = A + (size_t)(row0 + mr) * Kk;
  const float* Brow = TRANSB ? (B + (size_t)col * Kk) : B;

  v8f acc = {};
  for (int kk = 0; kk < Kk; kk += 64) {
    v16h a0 = load_frag_rowmajor(Arow, kk, g);
    v16h b0 = TRANSB ? load_frag_rowmajor(Brow, kk, g)
                     : load_fragB_kmajor(B, Nn, col, kk, g);
    acc = __builtin_amdgcn_wmma_f32_16x16x32_f16(false, a0, false, b0,
                                                 (short)0, acc, false, false);
    v16h a1 = load_frag_rowmajor(Arow, kk + 32, g);
    v16h b1 = TRANSB ? load_frag_rowmajor(Brow, kk + 32, g)
                     : load_fragB_kmajor(B, Nn, col, kk + 32, g);
    acc = __builtin_amdgcn_wmma_f32_16x16x32_f16(false, a1, false, b1,
                                                 (short)0, acc, false, false);
  }

  // Epilogue: hoist lane-invariant pieces out of the per-element loop.
  float bv = bias ? bias[col] : 0.f;
  int rowBase = row0 + (g << 3);               // C/D: M = r + 8*(lane/16), N = lane%16
  float* Cp = C + (size_t)rowBase * Nn + col;
  const float* Rp = resid ? (resid + (size_t)rowBase * Nn + col) : nullptr;
#pragma unroll
  for (int r = 0; r < 8; ++r) {
    float v = acc[r] + bv;
    if (doRelu) v = fmaxf(v, 0.f);
    if (Rp) v += Rp[(size_t)r * Nn];
    Cp[(size_t)r * Nn] = v;
  }
}

// ---------------------------------------------------------------------------
// Embedding gather + causal conv (K=3) + bias + ReLU -> conv_t (N*T, H)
// ---------------------------------------------------------------------------
__global__ void conv_kernel(const int* __restrict__ text,
                            const float* __restrict__ emb,
                            const float* __restrict__ w,   // (H_out, H_in, K)
                            const float* __restrict__ b,
                            float* __restrict__ out) {
  int r = blockIdx.x;            // n*T + t
  int n = r / TT, t = r % TT;
  int tid = threadIdx.x;         // 0..255
  __shared__ float xs[KW][HH];
  for (int k = 0; k < KW; ++k) {
    int tp = t + k - (KW - 1);   // input time index after causal pad
    float v = 0.f;
    if (tp >= 0) {
      int tok = (tp == 0) ? 0 : text[n * TT + tp - 1];  // shifted input
      v = emb[(size_t)tok * HH + tid];
    }
    xs[k][tid] = v;
  }
  __syncthreads();
  int o = tid;
  float acc = b[o];
  const float* wo = w + (size_t)o * HH * KW;
  for (int i = 0; i < HH; ++i) {
    acc += xs[0][i] * wo[i * KW + 0]
         + xs[1][i] * wo[i * KW + 1]
         + xs[2][i] * wo[i * KW + 2];
  }
  out[(size_t)r * HH + o] = fmaxf(acc, 0.f);
}

// Per-row max of logits + gather observed-token logit.  One block per row.
__global__ void rowmax_kernel(const float* __restrict__ logits,
                              const int* __restrict__ text,
                              float* __restrict__ lmax,
                              float* __restrict__ obslogit) {
  int r = blockIdx.x, tid = threadIdx.x;
  __shared__ float red[256];
  float m = -1e30f;
  for (int v = tid; v < VV; v += 256) m = fmaxf(m, logits[(size_t)r * VV + v]);
  red[tid] = m; __syncthreads();
  for (int s = 128; s > 0; s >>= 1) {
    if (tid < s) red[tid] = fmaxf(red[tid], red[tid + s]);
    __syncthreads();
  }
  if (tid == 0) {
    lmax[r] = red[0];
    obslogit[r] = logits[(size_t)r * VV + text[r]];
  }
}

// logits -> P = exp(logits - rowmax) in place.
__global__ void exp_kernel(float* __restrict__ P, const float* __restrict__ lmax) {
  int i = blockIdx.x * blockDim.x + threadIdx.x;
  if (i < NB * TT * VV) {
    int r = i / VV;
    P[i] = expf(P[i] - lmax[r]);
  }
}

// Per-state max of em_cv (C,V) + write E_vc[v][c] = exp(em_cv[c][v]-max).
__global__ void emax_kernel(const float* __restrict__ em_cv,
                            float* __restrict__ emax,
                            float* __restrict__ E_vc) {
  int c = blockIdx.x, tid = threadIdx.x;
  __shared__ float red[256];
  float m = -1e30f;
  for (int v = tid; v < VV; v += 256) m = fmaxf(m, em_cv[(size_t)c * VV + v]);
  red[tid] = m; __syncthreads();
  for (int s = 128; s > 0; s >>= 1) {
    if (tid < s) red[tid] = fmaxf(red[tid], red[tid + s]);
    __syncthreads();
  }
  m = red[0];
  if (tid == 0) emax[c] = m;
  for (int v = tid; v < VV; v += 256)
    E_vc[(size_t)v * CC + c] = expf(em_cv[(size_t)c * VV + v] - m);
}

// obs_em[r,c] = obslogit[r] + em_cv[c, tok(r)] - (lmax[r] + emax[c] + log(Zsum[r,c]))
__global__ void poe_kernel(const float* __restrict__ Zsum,
                           const float* __restrict__ lmax,
                           const float* __restrict__ emax,
                           const float* __restrict__ obslogit,
                           const float* __restrict__ em_cv,
                           const int* __restrict__ text,
                           float* __restrict__ obs_em) {
  int i = blockIdx.x * blockDim.x + threadIdx.x;
  if (i >= NB * TT * CC) return;
  int r = i / CC, c = i % CC;
  int tok = text[r];
  float logZ = lmax[r] + emax[c] + logf(Zsum[i]);
  obs_em[i] = obslogit[r] + em_cv[(size_t)c * VV + tok] - logZ;
}

// start[c] = log_softmax_c( sm_h @ sm_wo + sm_bo ).  One block, 64 threads.
__global__ void start_kernel(const float* __restrict__ sm_h,
                             const float* __restrict__ sm_wo,
                             const float* __restrict__ sm_bo,
                             float* __restrict__ start) {
  int c = threadIdx.x;  // 0..63
  __shared__ float red[CC];
  float acc = sm_bo[0];
  for (int k = 0; k < HH; ++k) acc += sm_h[(size_t)c * HH + k] * sm_wo[k];
  red[c] = acc; __syncthreads();
  for (int s = 32; s > 0; s >>= 1) {
    if (c < s) red[c] = fmaxf(red[c], red[c + s]);
    __syncthreads();
  }
  float m = red[0]; __syncthreads();
  red[c] = expf(acc - m); __syncthreads();
  for (int s = 32; s > 0; s >>= 1) {
    if (c < s) red[c] += red[c + s];
    __syncthreads();
  }
  start[c] = acc - (m + logf(red[0]));
}

// trans[cn][cp] = log_softmax over cn of pre[cp][cn], transposed on store.
__global__ void trans_kernel(const float* __restrict__ pre, float* __restrict__ trans) {
  int cp = blockIdx.x, cn = threadIdx.x;  // 64 blocks x 64 threads
  __shared__ float red[CC];
  float v = pre[(size_t)cp * CC + cn];
  red[cn] = v; __syncthreads();
  for (int s = 32; s > 0; s >>= 1) {
    if (cn < s) red[cn] = fmaxf(red[cn], red[cn + s]);
    __syncthreads();
  }
  float m = red[0]; __syncthreads();
  red[cn] = expf(v - m); __syncthreads();
  for (int s = 32; s > 0; s >>= 1) {
    if (cn < s) red[cn] += red[cn + s];
    __syncthreads();
  }
  trans[(size_t)cn * CC + cp] = v - (m + logf(red[0]));
}

// Sequential HMM forward recursion + evidence.  One block, 64 threads (=C).
__global__ void forward_kernel(const float* __restrict__ start,
                               const float* __restrict__ trans,
                               const float* __restrict__ obs_em,
                               float* __restrict__ out) {
  int c = threadIdx.x;  // 0..63
  __shared__ float tr[CC * CC];
  __shared__ float alpha[CC], red[CC];
  for (int i = c; i < CC * CC; i += CC) tr[i] = trans[i];
  __syncthreads();
  float total = 0.f;
  for (int n = 0; n < NB; ++n) {
    alpha[c] = start[c] + obs_em[(size_t)(n * TT) * CC + c];
    __syncthreads();
    for (int t = 1; t < TT; ++t) {
      const float* trc = tr + (size_t)c * CC;  // trans[c_next=c][c_prev]
      float m = -1e30f;
      for (int cp = 0; cp < CC; ++cp) m = fmaxf(m, alpha[cp] + trc[cp]);
      float s = 0.f;
      for (int cp = 0; cp < CC; ++cp) s += expf(alpha[cp] + trc[cp] - m);
      float a = m + logf(s) + obs_em[(size_t)(n * TT + t) * CC + c];
      __syncthreads();
      alpha[c] = a;
      __syncthreads();
    }
    red[c] = alpha[c]; __syncthreads();
    for (int s2 = 32; s2 > 0; s2 >>= 1) {
      if (c < s2) red[c] = fmaxf(red[c], red[c + s2]);
      __syncthreads();
    }
    float m2 = red[0]; __syncthreads();
    red[c] = expf(alpha[c] - m2); __syncthreads();
    for (int s2 = 32; s2 > 0; s2 >>= 1) {
      if (c < s2) red[c] += red[c + s2];
      __syncthreads();
    }
    if (c == 0) total += m2 + logf(red[0]);
    __syncthreads();
  }
  if (c == 0) out[0] = total;
}

// ---------------------------------------------------------------------------
extern "C" void kernel_launch(void* const* d_in, const int* in_sizes, int n_in,
                              void* d_out, int out_size, void* d_ws, size_t ws_size,
                              hipStream_t stream) {
  (void)in_sizes; (void)n_in; (void)out_size; (void)ws_size;
  const int*   text      = (const int*)  d_in[0];
  const float* emb       = (const float*)d_in[1];
  const float* cnn_w     = (const float*)d_in[2];
  const float* cnn_b     = (const float*)d_in[3];
  const float* ffr_w1    = (const float*)d_in[4];
  const float* ffr_b1    = (const float*)d_in[5];
  const float* ffr_w2    = (const float*)d_in[6];
  const float* ffr_b2    = (const float*)d_in[7];
  const float* proj_w    = (const float*)d_in[8];
  const float* start_emb = (const float*)d_in[9];
  const float* sm_w1     = (const float*)d_in[10];
  const float* sm_b1     = (const float*)d_in[11];
  const float* sm_w2     = (const float*)d_in[12];
  const float* sm_b2     = (const float*)d_in[13];
  const float* sm_wo     = (const float*)d_in[14];
  const float* sm_bo     = (const float*)d_in[15];
  const float* state_emb = (const float*)d_in[16];
  const float* tm_w1     = (const float*)d_in[17];
  const float* tm_b1     = (const float*)d_in[18];
  const float* tm_w2     = (const float*)d_in[19];
  const float* tm_b2     = (const float*)d_in[20];
  const float* tm_wo     = (const float*)d_in[21];
  const float* tm_bo     = (const float*)d_in[22];
  const float* pre_emb   = (const float*)d_in[23];
  const float* em_w1     = (const float*)d_in[24];
  const float* em_b1     = (const float*)d_in[25];
  const float* em_w2     = (const float*)d_in[26];
  const float* em_b2     = (const float*)d_in[27];
  const float* em_wo     = (const float*)d_in[28];
  const float* em_bo     = (const float*)d_in[29];

  const int M = NB * TT;  // 384

  // Workspace bump allocator (floats).
  float* W = (float*)d_ws;
  size_t off = 0;
  auto alloc = [&](size_t n) { float* p = W + off; off += n; return p; };
  float* conv_t    = alloc((size_t)M * HH);     // (384,256)
  float* t1        = alloc((size_t)M * HH);     // scratch MLP hidden
  float* h         = alloc((size_t)M * HH);     // residual MLP output
  float* logits    = alloc((size_t)M * VV);     // (384,4096); becomes P in place
  float* sm_h      = alloc((size_t)CC * HH);
  float* tm_h      = alloc((size_t)CC * HH);
  float* em_h      = alloc((size_t)CC * HH);
  float* trans_pre = alloc((size_t)CC * CC);
  float* trans     = alloc((size_t)CC * CC);
  float* em_cv     = alloc((size_t)CC * VV);    // (C,V)
  float* E_vc      = alloc((size_t)VV * CC);    // (V,C)
  float* lmax      = alloc(M);
  float* obslogit  = alloc(M);
  float* emax      = alloc(CC);
  float* Zsum      = alloc((size_t)M * CC);
  float* obs_em    = alloc((size_t)M * CC);
  float* startv    = alloc(CC);

  auto gemm = [&](const float* A, const float* B, const float* bias,
                  const float* resid, float* C, int Mm, int Nn, int Kk,
                  int transB, int relu) {
    int tiles = (Mm / 16) * (Nn / 16);
    int blocks = (tiles + 7) / 8;               // 8 waves (256 threads) / block
    if (transB)
      wmma_gemm_kernel<1><<<blocks, 256, 0, stream>>>(A, B, bias, resid, C,
                                                      Mm, Nn, Kk, relu);
    else
      wmma_gemm_kernel<0><<<blocks, 256, 0, stream>>>(A, B, bias, resid, C,
                                                      Mm, Nn, Kk, relu);
  };

  // 1) embedding + causal conv + ReLU
  conv_kernel<<<M, HH, 0, stream>>>(text, emb, cnn_w, cnn_b, conv_t);

  // 2) ffr residual MLP: h = relu(relu(conv@w1+b1)@w2+b2) + conv
  gemm(conv_t, ffr_w1, ffr_b1, nullptr, t1, M, HH, HH, 0, 1);
  gemm(t1, ffr_w2, ffr_b2, conv_t, h, M, HH, HH, 0, 1);

  // 3) logits = h @ proj_w^T   (proj_w is (V,H) -> transB)
  gemm(h, proj_w, nullptr, nullptr, logits, M, VV, HH, 1, 0);

  // 4) small residual MLPs (C=64 rows each)
  gemm(start_emb, sm_w1, sm_b1, nullptr, t1, CC, HH, HH, 0, 1);
  gemm(t1, sm_w2, sm_b2, start_emb, sm_h, CC, HH, HH, 0, 1);
  gemm(state_emb, tm_w1, tm_b1, nullptr, t1, CC, HH, HH, 0, 1);
  gemm(t1, tm_w2, tm_b2, state_emb, tm_h, CC, HH, HH, 0, 1);
  gemm(pre_emb, em_w1, em_b1, nullptr, t1, CC, HH, HH, 0, 1);
  gemm(t1, em_w2, em_b2, pre_emb, em_h, CC, HH, HH, 0, 1);

  // 5) HMM parameter heads
  start_kernel<<<1, CC, 0, stream>>>(sm_h, sm_wo, sm_bo, startv);
  gemm(tm_h, tm_wo, tm_bo, nullptr, trans_pre, CC, CC, HH, 0, 0);
  trans_kernel<<<CC, CC, 0, stream>>>(trans_pre, trans);
  gemm(em_h, em_wo, em_bo, nullptr, em_cv, CC, VV, HH, 0, 0);

  // 6) PoE emission via max-normalized exp GEMM:
  //    logZ[r,c] = lmax[r] + emax[c] + log( P(384x4096) @ E(4096x64) )
  rowmax_kernel<<<M, 256, 0, stream>>>(logits, text, lmax, obslogit);
  exp_kernel<<<(M * VV + 255) / 256, 256, 0, stream>>>(logits, lmax);
  emax_kernel<<<CC, 256, 0, stream>>>(em_cv, emax, E_vc);
  gemm(logits /*=P*/, E_vc, nullptr, nullptr, Zsum, M, CC, VV, 0, 0);
  poe_kernel<<<(M * CC + 255) / 256, 256, 0, stream>>>(Zsum, lmax, emax, obslogit,
                                                       em_cv, text, obs_em);

  // 7) forward recursion + evidence -> scalar
  forward_kernel<<<1, CC, 0, stream>>>(startv, trans, obs_em, (float*)d_out);
}
